// LinearAttention_4286377361534
// MI455X (gfx1250) — matrix operand order repaired and checked by reading, not compiled, not run on hardware
//
#include <hip/hip_runtime.h>

// ---------------------------------------------------------------------------
// LinearAttention (RoPE + elu softmax attention + LePE) for MI455X / gfx1250.
// Matrix work on v_wmma_f32_16x16x32_f16; QKV GEMM weight panels staged into
// LDS by the Tensor Data Mover (tensor_load_to_lds + s_wait_tensorcnt),
// double-buffered one k-chunk ahead.
// ---------------------------------------------------------------------------

typedef __attribute__((ext_vector_type(16))) _Float16 v16h;
typedef __attribute__((ext_vector_type(8)))  _Float16 v8h;
typedef __attribute__((ext_vector_type(8)))  float    v8f;
typedef __attribute__((ext_vector_type(4)))  unsigned int v4u;
typedef __attribute__((ext_vector_type(8)))  int      v8i;
typedef __attribute__((ext_vector_type(4)))  int      v4i;

#define WMMA_F16(a, b, c) \
    __builtin_amdgcn_wmma_f32_16x16x32_f16(false, (a), false, (b), (short)0, (c), false, false)

constexpr int Bb = 32, Nn = 512, Cc = 1024, Hh = 16, Dd = 64;
constexpr int M_ROWS = Bb * Nn;   // 16384 rows of x
constexpr int NQKV   = 3 * Cc;    // 3072 output features
constexpr int Kdim   = Cc;        // 1024 reduction dim
constexpr int KC     = 128;       // k-chunk staged in LDS per step
constexpr int NCHUNK = Kdim / KC; // 8

union AFrag { v16h v; v8h h[2]; };

// --- Tensor Data Mover availability / arity ---------------------------------
#if defined(__has_builtin)
#  if __has_builtin(__builtin_amdgcn_tensor_load_to_lds)
#    define HAVE_TDM 1
#  endif
#endif
#if defined(__has_include)
#  if __has_include(<hip/amd_detail/amd_gfx1250_TDM.h>)
#    define TDM_ARITY6 1   /* therock-10.0 headers -> 6-arg builtin */
#  endif
#endif

#ifdef HAVE_TDM
// Issue a 2D TDM load: tile_d0 (contiguous, halves) x tile_d1 rows from a
// row-major f16 tensor (row stride stride0 halves) into LDS at lds_addr.
// D# bit layout per CDNA5 ISA ch.8 (group0: count/lds/global/type,
// group1: data_size=2B, dims, tile dims, dim0 stride).
__device__ inline void tdm_load_2d(unsigned lds_addr, const void* gaddr,
                                   unsigned tensor_d0, unsigned tensor_d1,
                                   unsigned tile_d0, unsigned tile_d1,
                                   unsigned long long stride0) {
    unsigned long long ga = (unsigned long long)gaddr;
    v4u g0 = { 1u,                                        // count=1, user D#
               lds_addr,                                  // LDS byte address
               (unsigned)(ga & 0xFFFFFFFFu),              // global_addr[31:0]
               (unsigned)((ga >> 32) & 0x1FFFFFFu) | (2u << 30) };  // [56:32]|type=2
    v8i g1 = { (int)(1u << 16),                           // data_size = 2 bytes
               (int)((tensor_d0 & 0xFFFFu) << 16),        // tensor_dim0[15:0]
               (int)((tensor_d0 >> 16) | ((tensor_d1 & 0xFFFFu) << 16)),
               (int)((tensor_d1 >> 16) | (tile_d0 << 16)),// tile_dim0
               (int)(tile_d1),                            // tile_dim1, tile_dim2=0
               (int)(stride0 & 0xFFFFFFFFull),            // dim0_stride[31:0]
               (int)((stride0 >> 32) & 0xFFFFull),        // dim0_stride[47:32]
               0 };
    v4i gz = { 0, 0, 0, 0 };
#ifdef TDM_ARITY6
    v8i gz8 = { 0, 0, 0, 0, 0, 0, 0, 0 };
    __builtin_amdgcn_tensor_load_to_lds(g0, g1, gz, gz, gz8, 0);
#else
    __builtin_amdgcn_tensor_load_to_lds(g0, g1, gz, gz, 0);
#endif
}
#endif

// ---------------------------------------------------------------------------
// Kernel 1: f32 -> f16 conversion (x and w_qkv)
// ---------------------------------------------------------------------------
__global__ void cvt_f32_f16(const float* __restrict__ src,
                            _Float16* __restrict__ dst, int n) {
    int i = blockIdx.x * blockDim.x + threadIdx.x;
    if (i < n) dst[i] = (_Float16)src[i];
}

// ---------------------------------------------------------------------------
// Kernel 2: QKV GEMM.  out[m, n] = sum_k x[m,k] * w_qkv[n,k] + b[n]
// Block: 8 waves, 256(M) x 64(N).  Wave: 32(M) x 64(N), K in 32-steps.
// Weight panel (64 cols x 128 k, f16, 16KB) staged in LDS by the TDM,
// double-buffered; A-frags streamed from global (contiguous K order, same as B:
// lane-half g holds K 16g..16g+15).
// ---------------------------------------------------------------------------
__global__ __launch_bounds__(256)
void qkv_gemm(const _Float16* __restrict__ xh,
              const _Float16* __restrict__ wh,
              const float* __restrict__ bias,
              float* __restrict__ qkv) {
    __shared__ _Float16 ldsB[2][64 * KC];   // 2 x 16KB double buffer

    const int wave = threadIdx.x >> 5;
    const int lane = threadIdx.x & 31;
    const int l    = lane & 15;
    const int half = lane >> 4;

    const int bm = blockIdx.x % (M_ROWS / 256);
    const int bn = blockIdx.x / (M_ROWS / 256);
    const int m0 = bm * 256 + wave * 32;
    const int n0 = bn * 64;

    const _Float16* arow0 = xh + (size_t)(m0 + l) * Kdim + half * 16;
    const _Float16* arow1 = arow0 + (size_t)16 * Kdim;

    v8f acc[2][4] = {};

#ifdef HAVE_TDM
    // Prologue: stage chunk 0 into buffer 0.
    if (wave == 0)
        tdm_load_2d((unsigned)(size_t)&ldsB[0][0],
                    wh + (size_t)n0 * Kdim,
                    Kdim, NQKV, KC, 64, (unsigned long long)Kdim);
#endif

    for (int c = 0; c < NCHUNK; ++c) {
        const int buf = c & 1;

#ifdef HAVE_TDM
        if (wave == 0) {
            if (c + 1 < NCHUNK) {
                // Issue next chunk into the other buffer (safe: it was last
                // read in the previous iteration, behind the closing barrier).
                tdm_load_2d((unsigned)(size_t)&ldsB[buf ^ 1][0],
                            wh + (size_t)n0 * Kdim + (size_t)(c + 1) * KC,
                            Kdim, NQKV, KC, 64, (unsigned long long)Kdim);
                __builtin_amdgcn_s_wait_tensorcnt(1);   // chunk c has landed
            } else {
                __builtin_amdgcn_s_wait_tensorcnt(0);
            }
        }
#else
        // Fallback: cooperative global->LDS copy of the 64 x KC panel.
        {
            const _Float16* src = wh + (size_t)n0 * Kdim + (size_t)c * KC;
            for (int i = threadIdx.x; i < 64 * KC / 8; i += 256) {
                const int rowL = i >> 4;       // KC/8 == 16 v8h per row
                const int col8 = i & 15;
                *(v8h*)&ldsB[buf][rowL * KC + col8 * 8] =
                    *(const v8h*)(src + (size_t)rowL * Kdim + col8 * 8);
            }
        }
#endif
        __syncthreads();   // panel visible to all 8 waves

#pragma unroll
        for (int ks = 0; ks < KC / 32; ++ks) {
            const int kg = c * KC + ks * 32;
            __builtin_prefetch(arow0 + kg + KC, 0, 0);   // global_prefetch_b8
            AFrag a0, a1;
            a0.h[0] = *(const v8h*)(arow0 + kg);
            a0.h[1] = *(const v8h*)(arow0 + kg + 8);
            a1.h[0] = *(const v8h*)(arow1 + kg);
            a1.h[1] = *(const v8h*)(arow1 + kg + 8);
#pragma unroll
            for (int t = 0; t < 4; ++t) {
                // B-frag: lane l = column n0+t*16+l, contiguous K in LDS row.
                const v16h bf =
                    *(const v16h*)&ldsB[buf][(t * 16 + l) * KC + ks * 32 + half * 16];
                acc[0][t] = WMMA_F16(a0.v, bf, acc[0][t]);
                acc[1][t] = WMMA_F16(a1.v, bf, acc[1][t]);
            }
        }
        __syncthreads();   // all waves done with buf before it is re-filled
    }

    const float bv0 = bias[n0 + l];
    const float bv1 = bias[n0 + 16 + l];
    const float bv2 = bias[n0 + 32 + l];
    const float bv3 = bias[n0 + 48 + l];

#pragma unroll
    for (int s = 0; s < 2; ++s) {
#pragma unroll
        for (int r = 0; r < 8; ++r) {
            const int row = m0 + s * 16 + r + 8 * half;  // C/D layout
            float* o = qkv + (size_t)row * NQKV + n0 + l;
            o[ 0] = acc[s][0][r] + bv0;
            o[16] = acc[s][1][r] + bv1;
            o[32] = acc[s][2][r] + bv2;
            o[48] = acc[s][3][r] + bv3;
        }
    }
}

// ---------------------------------------------------------------------------
// Kernel 3: elu+1 and RoPE on q,k; repack to head-major f16.
//   qh, kh : (B,H,N,D) f16   (q pre-scaled by 1/sqrt(D))
//   vhT    : (B,H,D,N) f16   (transposed so PV B-frags are contiguous)
//   vf     : (B,H,N,D) f32   (for the LePE depthwise conv)
// ---------------------------------------------------------------------------
__global__ void rope_epilogue(const float* __restrict__ qkv,
                              _Float16* __restrict__ qh,
                              _Float16* __restrict__ kh,
                              _Float16* __restrict__ vhT,
                              float* __restrict__ vf) {
    const long tid = (long)blockIdx.x * blockDim.x + threadIdx.x;
    const int j =  tid        & 31;        // rope pair index (0..31)
    const int n = (tid >> 5)  & (Nn - 1);
    const int h = (tid >> 14) & (Hh - 1);
    const int b =  (int)(tid >> 18);
    if (b >= Bb) return;

    const size_t base = ((size_t)(b * Nn + n)) * NQKV + h * Dd + 2 * j;
    float q1 = qkv[base + 0],          q2 = qkv[base + 1];
    float k1 = qkv[base + Cc + 0],     k2 = qkv[base + Cc + 1];
    float v1 = qkv[base + 2 * Cc + 0], v2 = qkv[base + 2 * Cc + 1];

    // elu(x)+1  ==  x+1 (x>0)  |  exp(x) (x<=0)
    q1 = q1 > 0.f ? q1 + 1.f : __expf(q1);
    q2 = q2 > 0.f ? q2 + 1.f : __expf(q2);
    k1 = k1 > 0.f ? k1 + 1.f : __expf(k1);
    k2 = k2 > 0.f ? k2 + 1.f : __expf(k2);

    // inv_freq = exp(2j * -ln(10000)/64)
    const float inv_freq = __expf(-0.28782313662425572f * (float)j);
    const float ang = (float)n * inv_freq;
    const float s = __sinf(ang), c = __cosf(ang);

    const float scale = 0.125f;                 // 1/sqrt(64) folded into q
    const float qa = (q1 * c - q2 * s) * scale;
    const float qb = (q1 * s + q2 * c) * scale;
    const float ka =  k1 * c - k2 * s;
    const float kb =  k1 * s + k2 * c;

    const size_t hd = ((size_t)((b * Hh + h) * Nn + n)) * Dd + 2 * j;
    qh[hd] = (_Float16)qa;  qh[hd + 1] = (_Float16)qb;
    kh[hd] = (_Float16)ka;  kh[hd + 1] = (_Float16)kb;
    vf[hd] = v1;            vf[hd + 1] = v2;

    const size_t tp = ((size_t)(b * Hh + h) * Dd + 2 * j) * Nn + n;
    vhT[tp]      = (_Float16)v1;
    vhT[tp + Nn] = (_Float16)v2;
}

// ---------------------------------------------------------------------------
// Kernel 4: attention.  128 threads (4 waves) per (b, h, 64-row tile).
//   Phase 1: S = q k^T  (WMMA) -> wave-private 16x512 f32 slice in LDS
//   Phase 2: row softmax (warp-shuffle reductions)
//   Phase 3: O = P v (WMMA), P cvt f32->f16 from LDS in-register
// Dynamic LDS 128 KB -> 2 workgroups per 320KB WGP.
// ---------------------------------------------------------------------------
__global__ __launch_bounds__(128)
void attn_kernel(const _Float16* __restrict__ qh,
                 const _Float16* __restrict__ kh,
                 const _Float16* __restrict__ vhT,
                 float* __restrict__ out) {
    extern __shared__ float smem[];
    const int wave = threadIdx.x >> 5;
    const int lane = threadIdx.x & 31;
    const int l    = lane & 15;
    const int half = lane >> 4;

    const int tile =  blockIdx.x       & 7;
    const int h    = (blockIdx.x >> 3) & 15;
    const int b    =  blockIdx.x >> 7;
    const size_t bh = (size_t)(b * Hh + h);
    const int m0 = tile * 64 + wave * 16;

    float* Sl = smem + wave * 16 * Nn;

    // ---- Phase 1: scores --------------------------------------------------
    const _Float16* qbase = qh + (bh * Nn + m0 + l) * Dd + half * 16;
    AFrag aq0, aq1;
    aq0.h[0] = *(const v8h*)(qbase);        aq0.h[1] = *(const v8h*)(qbase + 8);
    aq1.h[0] = *(const v8h*)(qbase + 32);   aq1.h[1] = *(const v8h*)(qbase + 40);

    const _Float16* kbase = kh + (bh * Nn + l) * Dd + half * 16;
    for (int nt = 0; nt < Nn / 16; ++nt) {
        const _Float16* kb = kbase + (size_t)nt * 16 * Dd;
        v8f s = {};
        s = WMMA_F16(aq0.v, *(const v16h*)(kb),      s);   // d = 0..31
        s = WMMA_F16(aq1.v, *(const v16h*)(kb + 32), s);   // d = 32..63
#pragma unroll
        for (int r = 0; r < 8; ++r)
            Sl[(r + 8 * half) * Nn + nt * 16 + l] = s[r];
    }
    __syncthreads();

    // ---- Phase 2: softmax over this wave's 16 rows ------------------------
    for (int row = 0; row < 16; ++row) {
        float* Sr = Sl + row * Nn;
        float m = -3.4e38f;
        for (int i = lane; i < Nn; i += 32) m = fmaxf(m, Sr[i]);
#pragma unroll
        for (int off = 16; off >= 1; off >>= 1) m = fmaxf(m, __shfl_xor(m, off, 32));
        float sum = 0.f;
        for (int i = lane; i < Nn; i += 32) {
            float e = __expf(Sr[i] - m);
            Sr[i] = e;
            sum += e;
        }
#pragma unroll
        for (int off = 16; off >= 1; off >>= 1) sum += __shfl_xor(sum, off, 32);
        const float inv = 1.f / sum;
        for (int i = lane; i < Nn; i += 32) Sr[i] *= inv;
    }
    __syncthreads();

    // ---- Phase 3: O = P @ V ----------------------------------------------
    v8f o0 = {}, o1 = {}, o2 = {}, o3 = {};
    const _Float16* vb = vhT + bh * Dd * Nn + (size_t)l * Nn;
    for (int kt = 0; kt < Nn / 32; ++kt) {
        const float* pr = Sl + l * Nn + kt * 32;
        v16h a;
#pragma unroll
        for (int jj = 0; jj < 8; ++jj) a[jj]     = (_Float16)pr[half * 16 + jj];
#pragma unroll
        for (int jj = 0; jj < 8; ++jj) a[8 + jj] = (_Float16)pr[half * 16 + 8 + jj];
        const int kc = kt * 32 + half * 16;
        o0 = WMMA_F16(a, *(const v16h*)(vb + 0 * 16 * Nn + kc), o0);
        o1 = WMMA_F16(a, *(const v16h*)(vb + 1 * 16 * Nn + kc), o1);
        o2 = WMMA_F16(a, *(const v16h*)(vb + 2 * 16 * Nn + kc), o2);
        o3 = WMMA_F16(a, *(const v16h*)(vb + 3 * 16 * Nn + kc), o3);
    }

#pragma unroll
    for (int r = 0; r < 8; ++r) {
        const int row = m0 + r + 8 * half;
        float* op = out + ((size_t)(b * Nn) + row) * Cc + h * Dd + l;
        op[ 0] = o0[r];
        op[16] = o1[r];
        op[32] = o2[r];
        op[48] = o3[r];
    }
}

// ---------------------------------------------------------------------------
// Kernel 5: LePE depthwise conv-3 + bias, added into attention output.
// ---------------------------------------------------------------------------
__global__ void lepe_add(const float* __restrict__ vf,
                         const float* __restrict__ w_lepe,
                         const float* __restrict__ b_lepe,
                         float* __restrict__ out) {
    const long tid = (long)blockIdx.x * blockDim.x + threadIdx.x;
    if (tid >= (long)Bb * Nn * Cc) return;
    const int  c = (int)(tid % Cc);
    const long r = tid / Cc;
    const int  n = (int)(r % Nn);
    const int  b = (int)(r / Nn);
    const int  h = c >> 6, d = c & 63;

    const float* vcol = vf + ((size_t)(b * Hh + h) * Nn) * Dd + d;
    const float* wl = w_lepe + c * 3;
    float acc = b_lepe[c];
    if (n > 0)      acc += vcol[(size_t)(n - 1) * Dd] * wl[0];
                    acc += vcol[(size_t) n      * Dd] * wl[1];
    if (n < Nn - 1) acc += vcol[(size_t)(n + 1) * Dd] * wl[2];
    out[tid] += acc;
}

// ---------------------------------------------------------------------------
// Launch
// ---------------------------------------------------------------------------
extern "C" void kernel_launch(void* const* d_in, const int* in_sizes, int n_in,
                              void* d_out, int out_size, void* d_ws, size_t ws_size,
                              hipStream_t stream) {
    (void)in_sizes; (void)n_in; (void)out_size; (void)ws_size;

    const float* x      = (const float*)d_in[0];
    const float* w_qkv  = (const float*)d_in[1];
    const float* b_qkv  = (const float*)d_in[2];
    const float* w_lepe = (const float*)d_in[3];
    const float* b_lepe = (const float*)d_in[4];
    float* out = (float*)d_out;

    char*  ws  = (char*)d_ws;
    size_t off = 0;
    auto alloc = [&](size_t bytes) -> void* {
        void* p = ws + off;
        off += (bytes + 255) & ~(size_t)255;
        return p;
    };

    const size_t headElems = (size_t)Bb * Hh * Nn * Dd;     // 16.7M
    _Float16* xh  = (_Float16*)alloc((size_t)M_ROWS * Kdim * 2);
    _Float16* wh  = (_Float16*)alloc((size_t)NQKV   * Kdim * 2);
    float*    qkv = (float*)   alloc((size_t)M_ROWS * NQKV * 4);
    _Float16* qh  = (_Float16*)alloc(headElems * 2);
    _Float16* kh  = (_Float16*)alloc(headElems * 2);
    _Float16* vhT = (_Float16*)alloc(headElems * 2);
    float*    vf  = (float*)   alloc(headElems * 4);

    // 1) f32 -> f16 converts
    {
        int n = M_ROWS * Kdim;
        cvt_f32_f16<<<(n + 255) / 256, 256, 0, stream>>>(x, xh, n);
        n = NQKV * Kdim;
        cvt_f32_f16<<<(n + 255) / 256, 256, 0, stream>>>(w_qkv, wh, n);
    }

    // 2) QKV GEMM (16384 x 3072 x 1024): TDM-staged B panels + WMMA
    qkv_gemm<<<(M_ROWS / 256) * (NQKV / 64), 256, 0, stream>>>(xh, wh, b_qkv, qkv);

    // 3) elu+1, RoPE, head-major repack
    {
        long threads = (long)Bb << 18;                       // B*H*N*D/2
        rope_epilogue<<<(unsigned)(threads / 256), 256, 0, stream>>>(qkv, qh, kh, vhT, vf);
    }

    // 4) attention: 4096 workgroups, 128 threads, 128KB dynamic LDS
    attn_kernel<<<Bb * Hh * (Nn / 64), 128, 4 * 16 * Nn * sizeof(float), stream>>>(
        qh, kh, vhT, out);

    // 5) LePE depthwise conv + add
    {
        long n = (long)Bb * Nn * Cc;
        lepe_add<<<(unsigned)((n + 255) / 256), 256, 0, stream>>>(vf, w_lepe, b_lepe, out);
    }
}
